// CoreAttention_30906584662632
// MI455X (gfx1250) — matrix operand order repaired and checked by reading, not compiled, not run on hardware
//
#include <hip/hip_runtime.h>
#include <hip/hip_bf16.h>
#include <math.h>

// ---------------------------------------------------------------------------
// Flash-attention for CDNA5 (gfx1250), wave32, bf16 WMMA with f32 accumulate.
// sq=2048, b=2, nh=16, hd=128, fp32 tensors, bool mask [b,1,sq,sq].
// Double-buffered K/V staging: global loads for block kb+1 are issued into
// registers while block kb is computed from LDS; one barrier per iteration.
// ---------------------------------------------------------------------------

typedef __bf16 bf16;
typedef __attribute__((ext_vector_type(16))) bf16  v16bf;
typedef __attribute__((ext_vector_type(8)))  bf16  v8bf;
typedef __attribute__((ext_vector_type(4)))  bf16  v4bf;
typedef __attribute__((ext_vector_type(8)))  float v8f;

namespace {
constexpr int SQ   = 2048;
constexpr int NB   = 2;     // batch
constexpr int NH   = 16;    // heads
constexpr int HD   = 128;   // head dim
constexpr int KBLK = 32;    // keys per inner iteration
constexpr int QWG  = 128;   // queries per workgroup (8 waves x 16 rows)
constexpr int NWAVE = QWG / 16;
constexpr int NCHUNK = (KBLK * HD / 4) / 256;  // float4 chunks per thread = 4
// net scale = COEFF / (sqrt(HD)*LAYER) = 2 / (2*sqrt(128)) = 1/sqrt(128)
constexpr float SCALE     = 0.0883883476483184f;
constexpr float MASK_FILL = -10000.0f;
}

__device__ __forceinline__ v16bf cat8(v8bf lo, v8bf hi) {
  v16bf r;
#pragma unroll
  for (int e = 0; e < 8; ++e) { r[e] = lo[e]; r[8 + e] = hi[e]; }
  return r;
}

__global__ __launch_bounds__(256, 1)
void flash_attn_bf16wmma(const float* __restrict__ Q,
                         const float* __restrict__ K,
                         const float* __restrict__ V,
                         const unsigned char* __restrict__ Mask,
                         float* __restrict__ Out) {
  // Double-buffered K (row-major bf16) and V^T (transposed bf16) + P slab.
  __shared__ __align__(16) bf16 sK [2][KBLK][HD];     // 2 x 8 KB
  __shared__ __align__(16) bf16 sVt[2][HD][KBLK];     // 2 x 8 KB
  __shared__ __align__(16) bf16 sP [NWAVE][16][KBLK]; // 8 KB

  const int tid   = (int)threadIdx.x;
  const int wave  = tid >> 5;
  const int lane  = tid & 31;
  const int lhalf = lane >> 4;   // 0: lanes 0-15, 1: lanes 16-31
  const int l16   = lane & 15;

  const int bh = (int)blockIdx.y;        // 0 .. NB*NH-1
  const int b  = bh / NH;
  const int h  = bh % NH;
  const int qw = (int)blockIdx.x * QWG + wave * 16;  // this wave's query base

  // Per-thread staging geometry (fixed across iterations).
  int skey[NCHUNK], sdd[NCHUNK];
  size_t gchunk[NCHUNK];
#pragma unroll
  for (int j = 0; j < NCHUNK; ++j) {
    const int i = tid + j * 256;         // 0..1023 float4 positions
    skey[j] = i >> 5;                    // (i*4) / 128
    sdd[j]  = (i & 31) * 4;              // (i*4) % 128
    gchunk[j] = (((size_t)skey[j] * NB + b) * NH + h) * HD + sdd[j];
  }
  const size_t kvStride = (size_t)KBLK * NB * NH * HD;  // advance 32 keys

  // ---- Load this wave's Q tile (16 x 128) as 4 bf16 A-fragments (K=32 each).
  // ISA A layout (16-bit, 16x32): lane m / m+16 holds row m, runs of 8:
  //   elems[0..7]  = dims kc*32 + lhalf*8 + [0,8)
  //   elems[8..15] = dims kc*32 + 16 + lhalf*8 + [0,8)
  v16bf aQ[4];
  {
    const int qrow = qw + l16;
    const float* qp = Q + (((size_t)qrow * NB + b) * NH + h) * HD;
#pragma unroll
    for (int kc = 0; kc < 4; ++kc) {
      const int d0 = kc * 32 + lhalf * 8;
#pragma unroll
      for (int e = 0; e < 8; ++e) {
        aQ[kc][e]     = (bf16)qp[d0 + e];
        aQ[kc][8 + e] = (bf16)qp[d0 + 16 + e];
      }
    }
  }

  // ---- Output accumulators: 8 C-tiles (16x16 f32) covering d = 0..127.
  v8f acc[8];
#pragma unroll
  for (int ot = 0; ot < 8; ++ot)
#pragma unroll
    for (int e = 0; e < 8; ++e) acc[ot][e] = 0.0f;

  // Online-softmax row state (C layout: lanes 0-15/VGPR r -> row r,
  // lanes 16-31/VGPR r -> row 8+r).
  float mrow[8], lrow[8], alive[8];
#pragma unroll
  for (int r = 0; r < 8; ++r) { mrow[r] = -INFINITY; lrow[r] = 0.0f; alive[r] = 0.0f; }

  const unsigned char* mbase = Mask + (size_t)b * SQ * SQ + (size_t)qw * SQ;

  // ---- Prologue: stage block 0 into buffer 0.
  {
#pragma unroll
    for (int j = 0; j < NCHUNK; ++j) {
      const float4 kv = *(const float4*)(K + gchunk[j]);
      const float4 vv = *(const float4*)(V + gchunk[j]);
      v4bf kp = { (bf16)kv.x, (bf16)kv.y, (bf16)kv.z, (bf16)kv.w };
      *(v4bf*)&sK[0][skey[j]][sdd[j]] = kp;
      sVt[0][sdd[j] + 0][skey[j]] = (bf16)vv.x;
      sVt[0][sdd[j] + 1][skey[j]] = (bf16)vv.y;
      sVt[0][sdd[j] + 2][skey[j]] = (bf16)vv.z;
      sVt[0][sdd[j] + 3][skey[j]] = (bf16)vv.w;
    }
  }
  __syncthreads();

  for (int kb = 0; kb < SQ / KBLK; ++kb) {
    const int cur = kb & 1;
    const int nxt = cur ^ 1;
    const int kbase = kb * KBLK;
    const bool more = (kb + 1) < (SQ / KBLK);

    // ---- Issue global loads for block kb+1 into registers (in flight
    // while the WMMA chain below runs out of LDS buffer `cur`).
    float4 kreg[NCHUNK], vreg[NCHUNK];
    if (more) {
      const size_t base = (size_t)(kb + 1) * kvStride;
#pragma unroll
      for (int j = 0; j < NCHUNK; ++j) {
        kreg[j] = *(const float4*)(K + base + gchunk[j]);
        vreg[j] = *(const float4*)(V + base + gchunk[j]);
      }
    }

    // ---- S = Q * K^T for two 16-key N-tiles (scores for 32 keys).
    v8f sC[2];
#pragma unroll
    for (int nt = 0; nt < 2; ++nt) {
      v8f c;
#pragma unroll
      for (int e = 0; e < 8; ++e) c[e] = 0.0f;
      const int key = nt * 16 + l16;   // B column = key index (per lane)
#pragma unroll
      for (int kc = 0; kc < 4; ++kc) {
        // B layout (16-bit, 32x16): lane n holds col n, K = lhalf*8 + [0,8), 16 + lhalf*8 + [0,8)
        const bf16* kp = &sK[cur][key][kc * 32 + lhalf * 8];
        v16bf bK = cat8(*(const v8bf*)kp, *(const v8bf*)(kp + 16));
        c = __builtin_amdgcn_wmma_f32_16x16x32_bf16(
                false, aQ[kc], false, bK, (short)0, c, false, false);
      }
      sC[nt] = c;
    }

    // ---- Scale, mask-fill, row max (xor-shuffle reduce within 16 lanes).
    float rowmax[8];
#pragma unroll
    for (int r = 0; r < 8; ++r) {
      const unsigned char* mr =
          mbase + (size_t)(lhalf * 8 + r) * SQ + kbase + l16;
      const unsigned char m0 = mr[0];
      const unsigned char m1 = mr[16];
      alive[r] = fmaxf(alive[r], (m0 && m1) ? 0.0f : 1.0f);
      float x0 = m0 ? MASK_FILL : sC[0][r] * SCALE;
      float x1 = m1 ? MASK_FILL : sC[1][r] * SCALE;
      sC[0][r] = x0;
      sC[1][r] = x1;
      float lm = fmaxf(x0, x1);
      lm = fmaxf(lm, __shfl_xor(lm, 1, 16));
      lm = fmaxf(lm, __shfl_xor(lm, 2, 16));
      lm = fmaxf(lm, __shfl_xor(lm, 4, 16));
      lm = fmaxf(lm, __shfl_xor(lm, 8, 16));
      rowmax[r] = lm;
    }

    // ---- Online softmax update + rescale running output.
    float pe0[8], pe1[8], corr[8];
#pragma unroll
    for (int r = 0; r < 8; ++r) {
      const float mn = fmaxf(mrow[r], rowmax[r]);
      corr[r] = __expf(mrow[r] - mn);   // first iter: exp(-inf) = 0
      mrow[r] = mn;
      const float p0 = __expf(sC[0][r] - mn);
      const float p1 = __expf(sC[1][r] - mn);
      pe0[r] = p0; pe1[r] = p1;
      float ls = p0 + p1;
      ls += __shfl_xor(ls, 1, 16);
      ls += __shfl_xor(ls, 2, 16);
      ls += __shfl_xor(ls, 4, 16);
      ls += __shfl_xor(ls, 8, 16);
      lrow[r] = lrow[r] * corr[r] + ls;
    }
#pragma unroll
    for (int ot = 0; ot < 8; ++ot)
#pragma unroll
      for (int r = 0; r < 8; ++r) acc[ot][r] *= corr[r];

    // ---- P (C layout) -> A layout via per-wave LDS slab (same-wave LDS
    // ops are in-order, no barrier needed).
#pragma unroll
    for (int r = 0; r < 8; ++r) {
      const int prow = lhalf * 8 + r;
      sP[wave][prow][l16]      = (bf16)pe0[r];
      sP[wave][prow][16 + l16] = (bf16)pe1[r];
    }
    const bf16* pr = &sP[wave][l16][0];
    v16bf aP = cat8(*(const v8bf*)(pr + lhalf * 8),
                    *(const v8bf*)(pr + 16 + lhalf * 8));

    // ---- O += P * V  (K = 32 keys, 8 output N-tiles across d=0..127).
#pragma unroll
    for (int ot = 0; ot < 8; ++ot) {
      const bf16* vp = &sVt[cur][ot * 16 + l16][lhalf * 8];
      v16bf bV = cat8(*(const v8bf*)vp, *(const v8bf*)(vp + 16));
      acc[ot] = __builtin_amdgcn_wmma_f32_16x16x32_bf16(
                    false, aP, false, bV, (short)0, acc[ot], false, false);
    }

    // ---- Drain the in-flight loads: convert + store block kb+1 into `nxt`.
    if (more) {
#pragma unroll
      for (int j = 0; j < NCHUNK; ++j) {
        v4bf kp = { (bf16)kreg[j].x, (bf16)kreg[j].y,
                    (bf16)kreg[j].z, (bf16)kreg[j].w };
        *(v4bf*)&sK[nxt][skey[j]][sdd[j]] = kp;
        sVt[nxt][sdd[j] + 0][skey[j]] = (bf16)vreg[j].x;
        sVt[nxt][sdd[j] + 1][skey[j]] = (bf16)vreg[j].y;
        sVt[nxt][sdd[j] + 2][skey[j]] = (bf16)vreg[j].z;
        sVt[nxt][sdd[j] + 3][skey[j]] = (bf16)vreg[j].w;
      }
    }
    __syncthreads();   // one barrier per iteration
  }

  // ---- Normalize (zero fully-masked rows, like the reference) and store.
  float invr[8];
#pragma unroll
  for (int r = 0; r < 8; ++r) {
    float a = alive[r];
    a = fmaxf(a, __shfl_xor(a, 1, 16));
    a = fmaxf(a, __shfl_xor(a, 2, 16));
    a = fmaxf(a, __shfl_xor(a, 4, 16));
    a = fmaxf(a, __shfl_xor(a, 8, 16));
    invr[r] = (a > 0.5f) ? (1.0f / lrow[r]) : 0.0f;
  }
#pragma unroll
  for (int ot = 0; ot < 8; ++ot) {
#pragma unroll
    for (int r = 0; r < 8; ++r) {
      const int row = qw + lhalf * 8 + r;
      const int d   = ot * 16 + l16;
      Out[((size_t)row * NB + b) * (NH * HD) + (size_t)h * HD + d] =
          acc[ot][r] * invr[r];
    }
  }
}

extern "C" void kernel_launch(void* const* d_in, const int* in_sizes, int n_in,
                              void* d_out, int out_size, void* d_ws, size_t ws_size,
                              hipStream_t stream) {
  (void)in_sizes; (void)n_in; (void)out_size; (void)d_ws; (void)ws_size;
  const float* Q = (const float*)d_in[0];
  const float* K = (const float*)d_in[1];
  const float* V = (const float*)d_in[2];
  const unsigned char* M = (const unsigned char*)d_in[3];  // bool mask, 1B/elem
  float* Out = (float*)d_out;

  dim3 grid(SQ / QWG, NB * NH);   // (16, 32)
  dim3 block(256);                // 8 wave32 waves
  hipLaunchKernelGGL(flash_attn_bf16wmma, grid, block, 0, stream,
                     Q, K, V, M, Out);
}
